// DynamicTransformerWithLocalEnhanceBlockNPEG_22170621182638
// MI455X (gfx1250) — matrix operand order repaired and checked by reading, not compiled, not run on hardware
//
#include <hip/hip_runtime.h>
#include <hip/hip_bf16.h>
#include <math.h>

// ---------------- constants ----------------
#define BATCH   16
#define NTOK    1025
#define NPAD    1056          // 33*32, key/query padding
#define HEADS   6
#define HDIM    64
#define DIM     384
#define QKVDIM  1152
#define MLPH    1536
#define KEEPN   717           // ceil(0.7*1024)
#define NKEPT   718           // cls + kept
#define NOUT    719           // + extra token
#define MPAD    720           // padded kept tokens (45*16)
#define MTILES1 66            // NPAD/16
#define NTILESA 65            // ceil(1025/16)
#define MTILES2 45            // MPAD/16

typedef __attribute__((ext_vector_type(16))) _Float16 v16h;
typedef __attribute__((ext_vector_type(8)))  float    v8f;

union F8 { v8f v; float f[8]; };
union H16 { v16h v; unsigned u[8]; };

// ---------------- WMMA fragment helpers (CDNA5 wave32 layouts) ----------------
// A (16x32 f16): lane(0..15)->row M, group g=lane>>4; VGPR v holds K pair
//   K/2 = 8*(v>>2) + 4*g + (v&3)
__device__ __forceinline__ v16h frag_a(const unsigned* s, int strideU) {
  int l = threadIdx.x & 31, r = l & 15, g = l >> 4;
  H16 f;
#pragma unroll
  for (int vv = 0; vv < 8; ++vv) {
    int k2 = 8 * (vv >> 2) + 4 * g + (vv & 3);
    f.u[vv] = s[r * strideU + k2];
  }
  return f.v;
}
// B (32x16 f16): lane -> column N=l&15; VGPR v holds K pair: K/2 = 8*(v>>2) + 4*g + (v&3)
__device__ __forceinline__ v16h frag_b(const unsigned* s, int strideU) {
  int l = threadIdx.x & 31, o = l & 15, g = l >> 4;
  H16 f;
#pragma unroll
  for (int vv = 0; vv < 8; ++vv) f.u[vv] = s[o * strideU + 8 * (vv >> 2) + 4 * g + (vv & 3)];
  return f.v;
}
__device__ __forceinline__ v8f wmma16(v16h a, v16h b, v8f c) {
  return __builtin_amdgcn_wmma_f32_16x16x32_f16(false, a, false, b, (short)0, c,
                                                false, false);
}

// One-wave GEMM strip: acc[4] (16x64, f32) += A[16,K] * W[64,K]^T
// Stages one 16x32 A tile + one 64x32 W tile per K-chunk, issues 4 WMMAs,
// and prefetches the next K-chunk while the current one computes.
__device__ __forceinline__ void gemm_strip(v8f acc[4], const unsigned* Ag, int ldaU,
                                           const unsigned* Wg, int ldwU, int Ku,
                                           unsigned* sA, unsigned* sB) {
  int lane = threadIdx.x & 31;
  for (int kk = 0; kk < Ku; kk += 16) {
#pragma unroll
    for (int i = 0; i < 8; ++i) {            // A tile: 16 rows x 16 uints
      int e = lane + i * 32;
      int r = e >> 4, c = e & 15;
      sA[e] = Ag[(size_t)r * ldaU + kk + c];
    }
#pragma unroll
    for (int i = 0; i < 32; ++i) {           // W tile: 64 rows x 16 uints
      int e = lane + i * 32;
      int r = e >> 4, c = e & 15;
      sB[e] = Wg[(size_t)r * ldwU + kk + c];
    }
    if (kk + 16 < Ku) {                      // prefetch next K-chunk
      __builtin_prefetch(&Ag[(size_t)(lane & 15) * ldaU + kk + 16], 0, 3);
      __builtin_prefetch(&Wg[(size_t)(lane * 2) * ldwU + kk + 16], 0, 3);
    }
    __syncthreads();
    v16h a = frag_a(sA, 16);
#pragma unroll
    for (int f = 0; f < 4; ++f)
      acc[f] = wmma16(a, frag_b(sB + f * 256, 16), acc[f]);
    __syncthreads();
  }
}

// ---------------- small helper kernels ----------------
__global__ void w2h_kernel(const float* __restrict__ w, _Float16* __restrict__ o,
                           int n) {
  int i = blockIdx.x * blockDim.x + threadIdx.x;
  if (i < n) o[i] = (_Float16)w[i];
}

__global__ void init_kernel(float* cls, int* kept, int n) {
  int i = blockIdx.x * blockDim.x + threadIdx.x;
  if (i < n) { cls[i] = 0.f; kept[i] = 0; }
}

// LayerNorm: one row per block (128 threads). Pads with zeros past VN.
__global__ void ln_kernel(const float* __restrict__ x, const float* __restrict__ w,
                          const float* __restrict__ b, _Float16* __restrict__ out,
                          int VN, int PN) {
  int blk = blockIdx.x, bi = blk / PN, n = blk % PN, t = threadIdx.x;
  __shared__ float red[128];
  _Float16* orow = out + ((size_t)bi * PN + n) * DIM;
  if (n >= VN) {
    for (int c = t; c < DIM; c += 128) orow[c] = (_Float16)0.f;
    return;
  }
  const float* xrow = x + ((size_t)bi * VN + n) * DIM;
  float s = 0.f;
  for (int c = t; c < DIM; c += 128) s += xrow[c];
  red[t] = s; __syncthreads();
  for (int o = 64; o > 0; o >>= 1) { if (t < o) red[t] += red[t + o]; __syncthreads(); }
  float mean = red[0] * (1.f / DIM); __syncthreads();
  float s2 = 0.f;
  for (int c = t; c < DIM; c += 128) { float d = xrow[c] - mean; s2 += d * d; }
  red[t] = s2; __syncthreads();
  for (int o = 64; o > 0; o >>= 1) { if (t < o) red[t] += red[t + o]; __syncthreads(); }
  float rs = rsqrtf(red[0] * (1.f / DIM) + 1e-5f);
  for (int c = t; c < DIM; c += 128)
    orow[c] = (_Float16)((xrow[c] - mean) * rs * w[c] + b[c]);
}

// ---------------- QKV GEMM (16x64 strip per wave, scatter to q/k/v) ----------
__global__ __launch_bounds__(32) void qkv_kernel(
    const _Float16* __restrict__ xn, const _Float16* __restrict__ wq,
    const float* __restrict__ bias, _Float16* __restrict__ q,
    _Float16* __restrict__ k, _Float16* __restrict__ v) {
  __shared__ unsigned sA[256], sB[1024];
  int blk = blockIdx.x;
  int cg = blk % (QKVDIM / 64); blk /= (QKVDIM / 64);
  int mt = blk % MTILES1; int bi = blk / MTILES1;
  const unsigned* Ag = (const unsigned*)(xn + ((size_t)bi * NPAD + mt * 16) * DIM);
  const unsigned* Wg = (const unsigned*)(wq + (size_t)(cg * 64) * DIM);
  v8f acc[4]; acc[0] = (v8f){}; acc[1] = (v8f){}; acc[2] = (v8f){}; acc[3] = (v8f){};
  gemm_strip(acc, Ag, DIM / 2, Wg, DIM / 2, DIM / 2, sA, sB);
  int lane = threadIdx.x & 31, g = lane >> 4, cl = lane & 15;
#pragma unroll
  for (int f = 0; f < 4; ++f) {
    F8 a; a.v = acc[f];
    int col = cg * 64 + f * 16 + cl;
    int which = col / DIM, rem = col - which * DIM, h = rem >> 6, d = rem & 63;
    _Float16* dst = which == 0 ? q : (which == 1 ? k : v);
    float scale = which == 0 ? 0.125f : 1.0f;
#pragma unroll
    for (int vv = 0; vv < 8; ++vv) {
      int n = mt * 16 + vv + 8 * g;
      float val = (a.f[vv] + bias[col]) * scale;
      dst[((size_t)(bi * HEADS + h) * NPAD + n) * HDIM + d] = (_Float16)val;
    }
  }
}

// ---------------- flash attention: one 16-row tile per (b,h) per wave ----------
__global__ __launch_bounds__(32) void attn_kernel(
    const _Float16* __restrict__ q, const _Float16* __restrict__ k,
    const _Float16* __restrict__ v, _Float16* __restrict__ out) {
  __shared__ unsigned squ[512];             // q tile 16x64
  __shared__ unsigned sku[1024];            // k chunk 32x64
  __shared__ __align__(16) _Float16 svh[64 * 32];   // v chunk transposed [d][m]
  __shared__ __align__(16) _Float16 sph[16 * 32];   // probs tile
  int blk = blockIdx.x;
  int nt = blk % NTILESA; blk /= NTILESA;
  int h = blk % HEADS; int bi = blk / HEADS;
  int lane = threadIdx.x, g = lane >> 4, cl = lane & 15;
  size_t bh = (size_t)(bi * HEADS + h) * NPAD;

  const unsigned* qg = (const unsigned*)(q + (bh + nt * 16) * HDIM);
#pragma unroll
  for (int i = 0; i < 16; ++i) { int e = lane + i * 32; squ[e] = qg[(e >> 5) * 32 + (e & 31)]; }
  __syncthreads();
  v16h aq0 = frag_a(squ, 32);
  v16h aq1 = frag_a(squ + 16, 32);

  float runm[8], runl[8];
  F8 o[4];
#pragma unroll
  for (int vv = 0; vv < 8; ++vv) { runm[vv] = -1e30f; runl[vv] = 0.f; }
#pragma unroll
  for (int f = 0; f < 4; ++f) o[f].v = (v8f){};

  const unsigned* kg = (const unsigned*)(k + bh * HDIM);
  const _Float16* vg = v + bh * HDIM;

  for (int mc = 0; mc < NPAD / 32; ++mc) {
    int mbase = mc * 32;
#pragma unroll
    for (int i = 0; i < 32; ++i) {            // stage K chunk (32x64)
      int e = lane + i * 32;
      sku[e] = kg[(size_t)(mbase + (e >> 5)) * 32 + (e & 31)];
    }
#pragma unroll
    for (int i = 0; i < 64; ++i) {            // stage V chunk transposed
      int e = lane + i * 32;
      int m = e >> 6, d = e & 63;
      svh[d * 32 + m] = vg[(size_t)(mbase + m) * HDIM + d];
    }
    if (mbase + 32 < NPAD) {                  // prefetch next key chunk
      __builtin_prefetch(&kg[(size_t)(mbase + 32 + lane) * 32], 0, 3);
      __builtin_prefetch(&vg[(size_t)(mbase + 32 + lane) * HDIM], 0, 3);
    }
    __syncthreads();
    F8 s0, s1; s0.v = (v8f){}; s1.v = (v8f){};
    s0.v = wmma16(aq0, frag_b(sku, 32), s0.v);
    s0.v = wmma16(aq1, frag_b(sku + 16, 32), s0.v);
    s1.v = wmma16(aq0, frag_b(sku + 16 * 32, 32), s1.v);
    s1.v = wmma16(aq1, frag_b(sku + 16 * 32 + 16, 32), s1.v);
    int col0 = mbase + cl, col1 = col0 + 16;
#pragma unroll
    for (int vv = 0; vv < 8; ++vv) {
      float a0 = (col0 < NTOK) ? s0.f[vv] : -1e30f;
      float a1 = (col1 < NTOK) ? s1.f[vv] : -1e30f;
      float mx = fmaxf(a0, a1);
      for (int mm = 1; mm < 16; mm <<= 1) mx = fmaxf(mx, __shfl_xor(mx, mm, 32));
      float nm = fmaxf(runm[vv], mx);
      float al = __expf(runm[vv] - nm);
      float p0 = __expf(a0 - nm), p1 = __expf(a1 - nm);
      float rs = p0 + p1;
      for (int mm = 1; mm < 16; mm <<= 1) rs += __shfl_xor(rs, mm, 32);
      runl[vv] = runl[vv] * al + rs; runm[vv] = nm;
      o[0].f[vv] *= al; o[1].f[vv] *= al; o[2].f[vv] *= al; o[3].f[vv] *= al;
      int row = vv + 8 * g;
      sph[row * 32 + cl] = (_Float16)p0;
      sph[row * 32 + 16 + cl] = (_Float16)p1;
    }
    __syncthreads();
    v16h ap = frag_a((const unsigned*)sph, 16);
#pragma unroll
    for (int f = 0; f < 4; ++f)
      o[f].v = wmma16(ap, frag_b((const unsigned*)svh + f * 256, 16), o[f].v);
    __syncthreads();
  }
#pragma unroll
  for (int vv = 0; vv < 8; ++vv) {
    int n = nt * 16 + vv + 8 * g;
    if (n < NTOK) {
      float inv = 1.f / runl[vv];
#pragma unroll
      for (int f = 0; f < 4; ++f)
        out[((size_t)bi * NPAD + n) * DIM + h * HDIM + f * 16 + cl] =
            (_Float16)(o[f].f[vv] * inv);
    }
  }
}

// ---------------- cls-attention row (softmax of q0·K per head, averaged) -------
__global__ void cls_kernel(const _Float16* __restrict__ q,
                           const _Float16* __restrict__ k,
                           float* __restrict__ cls) {
  int h = blockIdx.x % HEADS, bi = blockIdx.x / HEADS, t = threadIdx.x;
  __shared__ float sc[NPAD];
  __shared__ float q0s[HDIM];
  __shared__ float red[256];
  size_t bh = (size_t)(bi * HEADS + h) * NPAD;
  if (t < HDIM) q0s[t] = (float)q[bh * HDIM + t];
  __syncthreads();
  for (int m = t; m < NPAD; m += 256) {
    float s = -1e30f;
    if (m < NTOK) {
      s = 0.f;
      const _Float16* kr = k + (bh + m) * HDIM;
      for (int d = 0; d < HDIM; ++d) s += q0s[d] * (float)kr[d];
    }
    sc[m] = s;
  }
  __syncthreads();
  float lm = -1e30f;
  for (int m = t; m < NPAD; m += 256) lm = fmaxf(lm, sc[m]);
  red[t] = lm; __syncthreads();
  for (int o = 128; o > 0; o >>= 1) { if (t < o) red[t] = fmaxf(red[t], red[t + o]); __syncthreads(); }
  float mx = red[0]; __syncthreads();
  float ls = 0.f;
  for (int m = t; m < NPAD; m += 256) ls += __expf(sc[m] - mx);
  red[t] = ls; __syncthreads();
  for (int o = 128; o > 0; o >>= 1) { if (t < o) red[t] += red[t + o]; __syncthreads(); }
  float inv = 1.f / red[0];
  for (int m = t; m < NPAD; m += 256)
    if (m >= 1 && m < NTOK)
      atomicAdd(&cls[bi * (NTOK - 1) + m - 1],
                __expf(sc[m] - mx) * inv * (1.0f / HEADS));
}

// ---------------- proj GEMM + residual -> x_res (f32) ----------------
__global__ __launch_bounds__(32) void proj_kernel(
    const _Float16* __restrict__ ao, const _Float16* __restrict__ wp,
    const float* __restrict__ bias, const float* __restrict__ x,
    float* __restrict__ xr) {
  __shared__ unsigned sA[256], sB[1024];
  int blk = blockIdx.x;
  int cg = blk % (DIM / 64); blk /= (DIM / 64);
  int mt = blk % NTILESA; int bi = blk / NTILESA;
  const unsigned* Ag = (const unsigned*)(ao + ((size_t)bi * NPAD + mt * 16) * DIM);
  const unsigned* Wg = (const unsigned*)(wp + (size_t)(cg * 64) * DIM);
  v8f acc[4]; acc[0] = (v8f){}; acc[1] = (v8f){}; acc[2] = (v8f){}; acc[3] = (v8f){};
  gemm_strip(acc, Ag, DIM / 2, Wg, DIM / 2, DIM / 2, sA, sB);
  int lane = threadIdx.x & 31, g = lane >> 4, cl = lane & 15;
#pragma unroll
  for (int f = 0; f < 4; ++f) {
    F8 a; a.v = acc[f];
    int col = cg * 64 + f * 16 + cl;
#pragma unroll
    for (int vv = 0; vv < 8; ++vv) {
      int n = mt * 16 + vv + 8 * g;
      if (n < NTOK) {
        size_t idx = ((size_t)bi * NTOK + n) * DIM + col;
        xr[idx] = a.f[vv] + bias[col] + x[idx];
      }
    }
  }
}

// ---------------- top-k via bitonic sort (jax tie-break: lower index first) ----
__global__ __launch_bounds__(1024) void topk_kernel(const float* __restrict__ cls,
                                                    int* __restrict__ idxo,
                                                    int* __restrict__ kept) {
  int bi = blockIdx.x, t = threadIdx.x;
  __shared__ float val[1024];
  __shared__ int sidx[1024];
  val[t] = cls[bi * 1024 + t]; sidx[t] = t;
  __syncthreads();
  for (int kk = 2; kk <= 1024; kk <<= 1)
    for (int j = kk >> 1; j > 0; j >>= 1) {
      int ixj = t ^ j;
      if (ixj > t) {
        bool dir = ((t & kk) == 0);
        float v1 = val[t], v2 = val[ixj];
        int i1 = sidx[t], i2 = sidx[ixj];
        bool before = (v2 > v1) || (v2 == v1 && i2 < i1); // s[ixj] ranks first
        if (before == dir) { val[t] = v2; val[ixj] = v1; sidx[t] = i2; sidx[ixj] = i1; }
      }
      __syncthreads();
    }
  if (t < KEEPN) { idxo[bi * KEEPN + t] = sidx[t]; kept[bi * 1024 + sidx[t]] = 1; }
}

// ---------------- gather kept tokens -> x_new ----------------
__global__ void gather_kernel(const float* __restrict__ xr,
                              const int* __restrict__ idx,
                              float* __restrict__ xn) {
  int r = blockIdx.x % NKEPT, bi = blockIdx.x / NKEPT, t = threadIdx.x;
  int src = (r == 0) ? 0 : 1 + idx[bi * KEEPN + (r - 1)];
  const float* s = xr + ((size_t)bi * NTOK + src) * DIM;
  float* d = xn + ((size_t)bi * NKEPT + r) * DIM;
  for (int c = t; c < DIM; c += 128) d[c] = s[c];
}

// ---------------- fused extra token ----------------
__global__ void extra_kernel(const float* __restrict__ xr,
                             const float* __restrict__ cls,
                             const int* __restrict__ kept,
                             float* __restrict__ extra) {
  int bi = blockIdx.x, t = threadIdx.x; // 384 threads
  float acc = 0.f;
  for (int j = 0; j < 1024; ++j)
    if (!kept[bi * 1024 + j])
      acc += xr[((size_t)bi * NTOK + 1 + j) * DIM + t] * cls[bi * 1024 + j];
  extra[bi * DIM + t] = acc;
}

// ---------------- fc1 GEMM + exact GELU ----------------
__global__ __launch_bounds__(32) void fc1_kernel(
    const _Float16* __restrict__ xn2, const _Float16* __restrict__ w1,
    const float* __restrict__ bias, _Float16* __restrict__ hout) {
  __shared__ unsigned sA[256], sB[1024];
  int blk = blockIdx.x;
  int cg = blk % (MLPH / 64); blk /= (MLPH / 64);
  int mt = blk % MTILES2; int bi = blk / MTILES2;
  const unsigned* Ag = (const unsigned*)(xn2 + ((size_t)bi * MPAD + mt * 16) * DIM);
  const unsigned* Wg = (const unsigned*)(w1 + (size_t)(cg * 64) * DIM);
  v8f acc[4]; acc[0] = (v8f){}; acc[1] = (v8f){}; acc[2] = (v8f){}; acc[3] = (v8f){};
  gemm_strip(acc, Ag, DIM / 2, Wg, DIM / 2, DIM / 2, sA, sB);
  int lane = threadIdx.x & 31, g = lane >> 4, cl = lane & 15;
#pragma unroll
  for (int f = 0; f < 4; ++f) {
    F8 a; a.v = acc[f];
    int col = cg * 64 + f * 16 + cl;
#pragma unroll
    for (int vv = 0; vv < 8; ++vv) {
      int n = mt * 16 + vv + 8 * g;
      float xv = a.f[vv] + bias[col];
      float ge = 0.5f * xv * (1.f + erff(xv * 0.70710678118654752f));
      hout[((size_t)bi * MPAD + n) * MLPH + col] = (_Float16)ge;
    }
  }
}

// ---------------- fc2 GEMM + residual -> final output rows ----------------
__global__ __launch_bounds__(32) void fc2_kernel(
    const _Float16* __restrict__ hin, const _Float16* __restrict__ w2,
    const float* __restrict__ bias, const float* __restrict__ xnew,
    float* __restrict__ out) {
  __shared__ unsigned sA[256], sB[1024];
  int blk = blockIdx.x;
  int cg = blk % (DIM / 64); blk /= (DIM / 64);
  int mt = blk % MTILES2; int bi = blk / MTILES2;
  const unsigned* Ag = (const unsigned*)(hin + ((size_t)bi * MPAD + mt * 16) * MLPH);
  const unsigned* Wg = (const unsigned*)(w2 + (size_t)(cg * 64) * MLPH);
  v8f acc[4]; acc[0] = (v8f){}; acc[1] = (v8f){}; acc[2] = (v8f){}; acc[3] = (v8f){};
  gemm_strip(acc, Ag, MLPH / 2, Wg, MLPH / 2, MLPH / 2, sA, sB);
  int lane = threadIdx.x & 31, g = lane >> 4, cl = lane & 15;
#pragma unroll
  for (int f = 0; f < 4; ++f) {
    F8 a; a.v = acc[f];
    int col = cg * 64 + f * 16 + cl;
#pragma unroll
    for (int vv = 0; vv < 8; ++vv) {
      int n = mt * 16 + vv + 8 * g;
      if (n < NKEPT) {
        out[((size_t)bi * NOUT + n) * DIM + col] =
            a.f[vv] + bias[col] + xnew[((size_t)bi * NKEPT + n) * DIM + col];
      }
    }
  }
}

// ---------------- finalize: extra token row + n_tokens scalar ----------------
__global__ void final_kernel(const float* __restrict__ extra, float* __restrict__ out) {
  int bi = blockIdx.x, t = threadIdx.x;
  out[((size_t)bi * NOUT + NKEPT) * DIM + t] = extra[bi * DIM + t];
  if (bi == 0 && t == 0) out[(size_t)BATCH * NOUT * DIM] = (float)NKEPT;
}

// ---------------- host launch ----------------
extern "C" void kernel_launch(void* const* d_in, const int* in_sizes, int n_in,
                              void* d_out, int out_size, void* d_ws, size_t ws_size,
                              hipStream_t stream) {
  const float* x      = (const float*)d_in[0];
  const float* ln1w   = (const float*)d_in[1];
  const float* ln1b   = (const float*)d_in[2];
  const float* qkvw   = (const float*)d_in[3];
  const float* qkvb   = (const float*)d_in[4];
  const float* projw  = (const float*)d_in[5];
  const float* projb  = (const float*)d_in[6];
  const float* ln2w   = (const float*)d_in[7];
  const float* ln2b   = (const float*)d_in[8];
  const float* fc1w   = (const float*)d_in[9];
  const float* fc1b   = (const float*)d_in[10];
  const float* fc2w   = (const float*)d_in[11];
  const float* fc2b   = (const float*)d_in[12];
  float* out = (float*)d_out;

  char* ws = (char*)d_ws;
  size_t off = 0;
  auto take = [&](size_t bytes) {
    size_t o = off; off += (bytes + 255) & ~(size_t)255; return o;
  };
  _Float16* xn_h   = (_Float16*)(ws + take((size_t)BATCH * NPAD * DIM * 2));
  _Float16* wqkv_h = (_Float16*)(ws + take((size_t)QKVDIM * DIM * 2));
  _Float16* wproj_h= (_Float16*)(ws + take((size_t)DIM * DIM * 2));
  _Float16* wfc1_h = (_Float16*)(ws + take((size_t)MLPH * DIM * 2));
  _Float16* wfc2_h = (_Float16*)(ws + take((size_t)DIM * MLPH * 2));
  _Float16* qb     = (_Float16*)(ws + take((size_t)BATCH * HEADS * NPAD * HDIM * 2));
  _Float16* kb     = (_Float16*)(ws + take((size_t)BATCH * HEADS * NPAD * HDIM * 2));
  _Float16* vb     = (_Float16*)(ws + take((size_t)BATCH * HEADS * NPAD * HDIM * 2));
  _Float16* ao     = (_Float16*)(ws + take((size_t)BATCH * NPAD * DIM * 2));
  float*    xr     = (float*)   (ws + take((size_t)BATCH * NTOK * DIM * 4));
  float*    cls    = (float*)   (ws + take((size_t)BATCH * 1024 * 4));
  int*      tidx   = (int*)     (ws + take((size_t)BATCH * KEEPN * 4));
  int*      kept   = (int*)     (ws + take((size_t)BATCH * 1024 * 4));
  float*    xnew   = (float*)   (ws + take((size_t)BATCH * NKEPT * DIM * 4));
  _Float16* xn2_h  = (_Float16*)(ws + take((size_t)BATCH * MPAD * DIM * 2));
  _Float16* hh     = (_Float16*)(ws + take((size_t)BATCH * MPAD * MLPH * 2));
  float*    extra  = (float*)   (ws + take((size_t)BATCH * DIM * 4));

  // weight conversions fp32 -> fp16
  int nqkv = QKVDIM * DIM, nproj = DIM * DIM, nfc = MLPH * DIM;
  w2h_kernel<<<(nqkv + 255) / 256, 256, 0, stream>>>(qkvw, wqkv_h, nqkv);
  w2h_kernel<<<(nproj + 255) / 256, 256, 0, stream>>>(projw, wproj_h, nproj);
  w2h_kernel<<<(nfc + 255) / 256, 256, 0, stream>>>(fc1w, wfc1_h, nfc);
  w2h_kernel<<<(nfc + 255) / 256, 256, 0, stream>>>(fc2w, wfc2_h, nfc);

  // LN1 (zero-pads rows >= 1025)
  ln_kernel<<<BATCH * NPAD, 128, 0, stream>>>(x, ln1w, ln1b, xn_h, NTOK, NPAD);

  // QKV GEMM (16x64 strips)
  qkv_kernel<<<BATCH * MTILES1 * (QKVDIM / 64), 32, 0, stream>>>(
      xn_h, wqkv_h, qkvb, qb, kb, vb);

  // flash attention
  attn_kernel<<<BATCH * HEADS * NTILESA, 32, 0, stream>>>(qb, kb, vb, ao);

  // cls attention row + top-k
  init_kernel<<<(BATCH * 1024 + 255) / 256, 256, 0, stream>>>(cls, kept, BATCH * 1024);
  cls_kernel<<<BATCH * HEADS, 256, 0, stream>>>(qb, kb, cls);
  topk_kernel<<<BATCH, 1024, 0, stream>>>(cls, tidx, kept);

  // proj + residual
  proj_kernel<<<BATCH * NTILESA * (DIM / 64), 32, 0, stream>>>(ao, wproj_h, projb, x, xr);

  // gather + extra token
  gather_kernel<<<BATCH * NKEPT, 128, 0, stream>>>(xr, tidx, xnew);
  extra_kernel<<<BATCH, DIM, 0, stream>>>(xr, cls, kept, extra);

  // LN2 (zero-pads rows >= 718)
  ln_kernel<<<BATCH * MPAD, 128, 0, stream>>>(xnew, ln2w, ln2b, xn2_h, NKEPT, MPAD);

  // MLP
  fc1_kernel<<<BATCH * MTILES2 * (MLPH / 64), 32, 0, stream>>>(xn2_h, wfc1_h, fc1b, hh);
  fc2_kernel<<<BATCH * MTILES2 * (DIM / 64), 32, 0, stream>>>(hh, wfc2_h, fc2b, xnew, out);

  // extra token row + n_tokens
  final_kernel<<<BATCH, DIM, 0, stream>>>(extra, out);
}
